// ConvBlock_44427141710403
// MI455X (gfx1250) — matrix operand order repaired and checked by reading, not compiled, not run on hardware
//
#include <hip/hip_runtime.h>
#include <cstddef>
#include <cstdint>

// ---------- vector types ----------
typedef __attribute__((ext_vector_type(16))) __bf16 v16bf;
typedef __attribute__((ext_vector_type(8)))  __bf16 v8bf;
typedef __attribute__((ext_vector_type(4)))  __bf16 v4bf;
typedef __attribute__((ext_vector_type(8)))  float  v8f;
typedef __attribute__((ext_vector_type(4)))  float  v4f;

template <class A, class B> struct is_same_t { static constexpr bool value = false; };
template <class A> struct is_same_t<A, A>   { static constexpr bool value = true;  };

// ---------- bf16 <-> f32 helpers (RNE) ----------
__device__ __forceinline__ __bf16 f2bf(float f) {
    unsigned u; __builtin_memcpy(&u, &f, 4);
    unsigned r = (u + 0x7FFFu + ((u >> 16) & 1u)) >> 16;
    unsigned short s = (unsigned short)r;
    __bf16 b; __builtin_memcpy(&b, &s, 2);
    return b;
}
__device__ __forceinline__ float bf2f(__bf16 b) {
    unsigned short s; __builtin_memcpy(&s, &b, 2);
    unsigned u = ((unsigned)s) << 16;
    float f; __builtin_memcpy(&f, &u, 4);
    return f;
}
__device__ __forceinline__ float silu(float x) {
    return x / (1.0f + __expf(-x));
}

union V16U { v16bf v; v8bf h[2]; };

// ---------- CDNA5 async copy: global -> LDS, 16B per lane, ASYNCcnt ----------
__device__ __forceinline__ void async_copy_b128(uint32_t lds_byte_off,
                                                const __bf16* base,
                                                uint32_t glb_byte_off) {
    asm volatile("global_load_async_to_lds_b128 %0, %1, %2"
                 :: "v"(lds_byte_off), "v"(glb_byte_off), "s"(base)
                 : "memory");
}
__device__ __forceinline__ void wait_async0() {
    asm volatile("s_wait_asynccnt 0x0" ::: "memory");
}

// ---------- f32 -> bf16 bulk convert ----------
__global__ __launch_bounds__(256) void f32_to_bf16_kernel(const float* __restrict__ src,
                                                          __bf16* __restrict__ dst,
                                                          int n4) {
    int i = blockIdx.x * 256 + threadIdx.x;
    if (i < n4) {
        v4f v = ((const v4f*)src)[i];
        v4bf b;
#pragma unroll
        for (int j = 0; j < 4; ++j) b[j] = f2bf(v[j]);
        ((v4bf*)dst)[i] = b;
    }
}

// ---------- GEMM:  C[M,N] = A[M,K] * W[N,K]^T  (bf16 WMMA, f32 accum) ----------
// Block 256x128, 8 waves in a 4(M) x 2(N) grid, 64x64 tile per wave:
// 16 WMMAs per 16 ds_load_b128 per K-step -> 1.0 LDS loads per WMMA.
constexpr int BM = 256, BN = 128, BK = 32;
constexpr int LDT = BK + 8;              // padded LDS row stride: 80 B (16B-mult, conflict-free)
constexpr int TILE_HALVES = (BM + BN) * LDT;   // 15360 halves; x2 buffers = 60 KB LDS

// issue one block tile (A 256xBK + B 128xBK) as async b128 copies: 6 per thread
__device__ __forceinline__ void stage_tile(const __bf16* __restrict__ A,
                                           const __bf16* __restrict__ W,
                                           uint32_t aLds, int blockM, int blockN,
                                           int K, int k0, int t) {
    const uint32_t bLds = aLds + (uint32_t)(BM * LDT) * 2u;
#pragma unroll
    for (int it = 0; it < 4; ++it) {            // A: 1024 chunks / 256 threads
        int i = it * 256 + t;
        int r = i >> 2;                          // row 0..255
        int c = (i & 3) * 8;                     // col 0..24 step 8
        async_copy_b128(aLds + (uint32_t)(r * LDT + c) * 2u, A,
                        (uint32_t)(((blockM + r) * K + k0 + c) * 2));
    }
#pragma unroll
    for (int it = 0; it < 2; ++it) {            // B: 512 chunks / 256 threads
        int i = it * 256 + t;
        int r = i >> 2;                          // row 0..127
        int c = (i & 3) * 8;
        async_copy_b128(bLds + (uint32_t)(r * LDT + c) * 2u, W,
                        (uint32_t)(((blockN + r) * K + k0 + c) * 2));
    }
}

template <typename OT>
__global__ __launch_bounds__(256) void gemm_nt(const __bf16* __restrict__ A,
                                               const __bf16* __restrict__ W,
                                               OT* __restrict__ C,
                                               int M, int N, int K)
{
    __shared__ __attribute__((aligned(16))) __bf16 Sh[2][TILE_HALVES];

    const int t    = threadIdx.x;
    const int lane = t & 31;
    const int wid  = t >> 5;
    const int waveM = (wid & 3) * 64;   // 4 waves along M
    const int waveN = (wid >> 2) * 64;  // 2 waves along N
    const int blockN = blockIdx.x * BN;
    const int blockM = blockIdx.y * BM;

    v8f acc[4][4];
#pragma unroll
    for (int i = 0; i < 4; ++i)
#pragma unroll
        for (int j = 0; j < 4; ++j) acc[i][j] = v8f{};

    // ISA 7.12.2 fragment lane mapping
    const int arow  = lane & 15;          // A: row within 16-tile
    const int ahalf = (lane >> 4) * 8;    // A: K sub-offset 0 / 8 (and +16)
    const int bcol  = lane & 15;          // B: column
    const int bkoff = (lane >> 4) * 8;    // B: K sub-offset 0 / 8 (and +16)

    const int nk = K / BK;
    stage_tile(A, W, (uint32_t)(uintptr_t)&Sh[0][0], blockM, blockN, K, 0, t);

    for (int kt = 0; kt < nk; ++kt) {
        wait_async0();          // this thread's tile-kt copies have landed in LDS
        __syncthreads();        // ...and everyone else's too; prev reads of other buf done
        if (kt + 1 < nk)
            stage_tile(A, W, (uint32_t)(uintptr_t)&Sh[(kt + 1) & 1][0],
                       blockM, blockN, K, (kt + 1) * BK, t);

        const __bf16* As = &Sh[kt & 1][0];
        const __bf16* Bs = As + BM * LDT;

        v16bf afrag[4], bfrag[4];
#pragma unroll
        for (int fm = 0; fm < 4; ++fm) {
            int row = waveM + fm * 16 + arow;
            V16U u;
            u.h[0] = *(const v8bf*)(As + row * LDT + ahalf);
            u.h[1] = *(const v8bf*)(As + row * LDT + ahalf + 16);
            afrag[fm] = u.v;
        }
#pragma unroll
        for (int fn = 0; fn < 4; ++fn) {
            int col = waveN + fn * 16 + bcol;
            V16U u;
            u.h[0] = *(const v8bf*)(Bs + col * LDT + bkoff);
            u.h[1] = *(const v8bf*)(Bs + col * LDT + bkoff + 16);
            bfrag[fn] = u.v;
        }
#pragma unroll
        for (int fm = 0; fm < 4; ++fm)
#pragma unroll
            for (int fn = 0; fn < 4; ++fn)
                acc[fm][fn] = __builtin_amdgcn_wmma_f32_16x16x32_bf16(
                    false, afrag[fm], false, bfrag[fn],
                    (short)0, acc[fm][fn], false, false);
    }

    // ---- epilogue: C layout = lane(n), VGPR r -> m = r + 8*(lane>>4) ----
    const int mhalf = (lane >> 4) * 8;
    const int ncol  = lane & 15;
#pragma unroll
    for (int fm = 0; fm < 4; ++fm)
#pragma unroll
        for (int fn = 0; fn < 4; ++fn)
#pragma unroll
            for (int r = 0; r < 8; ++r) {
                int row = blockM + waveM + fm * 16 + r + mhalf;
                int col = blockN + waveN + fn * 16 + ncol;
                float v = acc[fm][fn][r];
                if constexpr (is_same_t<OT, __bf16>::value)
                    C[(size_t)row * N + col] = f2bf(v);
                else
                    C[(size_t)row * N + col] = v;
            }
}

// ---------- fused depthwise causal conv (K=4) + bias + silu*silu(gate) ----------
__global__ __launch_bounds__(256) void conv_gate_kernel(
    const __bf16* __restrict__ val,
    __bf16* __restrict__ gate,
    const float* __restrict__ Wc,
    const float* __restrict__ bc,
    int L)
{
    const int pos = blockIdx.x;        // b*L + l
    const int l   = pos % L;
    const int e0  = threadIdx.x * 8;   // 256 threads x 8 ch = 2048
    const size_t base = (size_t)pos * 2048 + e0;

    v8bf gb = *(const v8bf*)(gate + base);

    float w[8][4];
    const v4f* wp = (const v4f*)(Wc + (size_t)e0 * 4);
#pragma unroll
    for (int i = 0; i < 8; ++i) {
        v4f t4 = wp[i];
#pragma unroll
        for (int j = 0; j < 4; ++j) w[i][j] = t4[j];
    }
    v4f b0 = *(const v4f*)(bc + e0);
    v4f b1 = *(const v4f*)(bc + e0 + 4);

    float accv[8];
#pragma unroll
    for (int i = 0; i < 4; ++i) { accv[i] = b0[i]; accv[i + 4] = b1[i]; }

#pragma unroll
    for (int j = 0; j < 4; ++j) {
        int ls = l - 3 + j;
        if (ls >= 0) {
            v8bf vv = *(const v8bf*)(val + (size_t)(pos - 3 + j) * 2048 + e0);
#pragma unroll
            for (int i = 0; i < 8; ++i) accv[i] += w[i][j] * bf2f(vv[i]);
        }
    }

    v8bf outv;
#pragma unroll
    for (int i = 0; i < 8; ++i)
        outv[i] = f2bf(silu(accv[i]) * silu(bf2f(gb[i])));
    *(v8bf*)(gate + base) = outv;
}

// ---------- launch ----------
extern "C" void kernel_launch(void* const* d_in, const int* in_sizes, int n_in,
                              void* d_out, int out_size, void* d_ws, size_t ws_size,
                              hipStream_t stream)
{
    (void)in_sizes; (void)n_in; (void)out_size; (void)ws_size;

    const float* x  = (const float*)d_in[0];  // [B,L,1024]
    const float* Wp = (const float*)d_in[1];  // [2048,1024]
    const float* Wg = (const float*)d_in[2];  // [2048,1024]
    const float* Wc = (const float*)d_in[3];  // [2048,1,4]
    const float* bc = (const float*)d_in[4];  // [2048]
    const float* Wo = (const float*)d_in[5];  // [1024,2048]
    float* out = (float*)d_out;               // [B,L,1024]

    const int B = 2, L = 4096, D = 1024, E = 2048;
    const int M = B * L;                      // 8192

    // workspace layout (bf16 halves)
    __bf16* xb   = (__bf16*)d_ws;                 // [M, D]   16.8 MB
    __bf16* Wpb  = xb   + (size_t)M * D;          // [E, D]    4.2 MB
    __bf16* Wgb  = Wpb  + (size_t)E * D;          // [E, D]    4.2 MB
    __bf16* Wob  = Wgb  + (size_t)E * D;          // [D, E]    4.2 MB
    __bf16* val  = Wob  + (size_t)D * E;          // [M, E]   33.5 MB
    __bf16* gate = val  + (size_t)M * E;          // [M, E]   33.5 MB

    dim3 blk(256);

    // bulk f32 -> bf16 conversions (hoisted out of the GEMM inner loops)
    f32_to_bf16_kernel<<<dim3((M * D / 4 + 255) / 256), blk, 0, stream>>>(x,  xb,  M * D / 4);
    f32_to_bf16_kernel<<<dim3((E * D / 4 + 255) / 256), blk, 0, stream>>>(Wp, Wpb, E * D / 4);
    f32_to_bf16_kernel<<<dim3((E * D / 4 + 255) / 256), blk, 0, stream>>>(Wg, Wgb, E * D / 4);
    f32_to_bf16_kernel<<<dim3((D * E / 4 + 255) / 256), blk, 0, stream>>>(Wo, Wob, D * E / 4);

    // val = x @ Wp^T ; gate = x @ Wg^T
    gemm_nt<__bf16><<<dim3(E / BN, M / BM), blk, 0, stream>>>(xb, Wpb, val,  M, E, D);
    gemm_nt<__bf16><<<dim3(E / BN, M / BM), blk, 0, stream>>>(xb, Wgb, gate, M, E, D);
    // gate <- silu(conv(val)+b) * silu(gate)
    conv_gate_kernel<<<dim3(M), blk, 0, stream>>>(val, gate, Wc, bc, L);
    // out = gate @ Wo^T
    gemm_nt<float><<<dim3(D / BN, M / BM), blk, 0, stream>>>(gate, Wob, out, M, D, E);
}
